// VectorQuantizedVAE_48825188221335
// MI455X (gfx1250) — compile-verified, matrix operand order checked by reading, not run on hardware
//
#include <hip/hip_runtime.h>
#include <hip/hip_bf16.h>
#include <stdint.h>

typedef __attribute__((ext_vector_type(16))) _Float16     v16h;
typedef __attribute__((ext_vector_type(8)))  float        v8f;
typedef __attribute__((ext_vector_type(4)))  unsigned int v4u;
typedef __attribute__((ext_vector_type(8)))  unsigned int v8u;

struct U128 { unsigned long long a, b; };
union FragH { v16h v; _Float16 h[16]; U128 q[2]; };

#define BATCH 32
#define EPSBN 1e-5f

// ---------------------------------------------------------------------------
// Implicit-GEMM convolution (regular + stride-2 transposed) via WMMA f16.
// Block: 128 threads = 4 waves. Wave tile: 16 output pixels x 64 out channels
// (4 accumulators, shared A fragment -> 4 back-to-back WMMAs per K-chunk).
// Block tile: 64 pixels x 64 channels. Grid: (ceil(HW/64), ceil(Cout/64), B).
// A (64x32 f16) = im2col activations staged by VALU; weight chunk (64x32 f32)
// staged by the Tensor Data Mover (TDM) for normal convs (2D D# descriptor,
// OOB zero-fill handles K/Cout tails), VGPR-gather fallback for transposed.
// Bias folded into accumulator init.
// ---------------------------------------------------------------------------
__global__ void k_conv_wmma(const float* __restrict__ x, const float* __restrict__ w,
                            const float* __restrict__ bias, float* __restrict__ y,
                            int Cin, int Cout, int KH, int KW, int pad,
                            int Hin, int Win, int Hout, int Wout, int tmode) {
    __shared__ _Float16 Ash[64][32];   // 4 KB  im2col tile (f16)
    __shared__ _Float16 Bsh[64][32];   // 4 KB  weight tile (f16, n-major)
    __shared__ float    Wst[64][32];   // 8 KB  raw f32 weight staging (TDM target)

    const int tid  = threadIdx.x;
    const int wv   = tid >> 5;
    const int lane = tid & 31;
    const int lm   = lane & 15;
    const int lh   = lane >> 4;
    const int b    = blockIdx.z;
    const int co0  = blockIdx.y * 64;
    const int p0   = blockIdx.x * 64;
    const int HWo  = Hout * Wout;
    const int KK   = KH * KW;
    const int K    = Cin * KK;

    v8f c[4];
    #pragma unroll
    for (int ct = 0; ct < 4; ++ct) {
        const int co = co0 + ct * 16 + lm;
        const float bv = (co < Cout) ? bias[co] : 0.0f;
        #pragma unroll
        for (int r = 0; r < 8; ++r) c[ct][r] = bv;
    }

    for (int k0 = 0; k0 < K; k0 += 32) {
        // ---- stage im2col A tile: 64 pixels x 32 K ----
        for (int e = tid; e < 64 * 32; e += 128) {
            const int pl = e >> 5;
            const int kk = e & 31;
            const int k  = k0 + kk;
            float v = 0.0f;
            if (k < K) {
                const int ci  = k / KK;
                const int rem = k - ci * KK;
                const int kh  = rem / KW;
                const int kw  = rem - kh * KW;
                const int p   = p0 + pl;
                if (p < HWo) {
                    const int oy = p / Wout;
                    const int ox = p - oy * Wout;
                    int iy, ix; bool ok;
                    if (tmode) { // ConvTranspose2d k=4 s=2 p=1
                        const int ty = oy + 1 - kh;
                        const int tx = ox + 1 - kw;
                        ok = (ty >= 0) && (tx >= 0) && ((ty & 1) == 0) && ((tx & 1) == 0);
                        iy = ty >> 1; ix = tx >> 1;
                        ok = ok && (iy < Hin) && (ix < Win);
                    } else {
                        iy = oy + kh - pad; ix = ox + kw - pad;
                        ok = (iy >= 0) && (iy < Hin) && (ix >= 0) && (ix < Win);
                    }
                    if (ok) v = x[((b * Cin + ci) * Hin + iy) * Win + ix];
                }
            }
            Ash[pl][kk] = (_Float16)v;
        }

        // ---- stage raw weight chunk Wst[64][32] (f32) ----
        if (!tmode) {
            // TDM 2D tile: rows = out-channels (stride K), cols = 32 K-elems.
            // OOB reads (K tail, Cout tail) return zero per D# tensor dims.
            if (tid < 32) {
                const unsigned long long ga =
                    (unsigned long long)(uintptr_t)(w + (size_t)co0 * K + k0);
                const unsigned ldsoff = (unsigned)(uintptr_t)&Wst[0][0];
                const unsigned td0 = (unsigned)(K - k0);      // remaining cols
                const unsigned td1 = (unsigned)(Cout - co0);  // remaining rows
                v4u d0;
                d0[0] = 1u;                                   // count=1, user mode
                d0[1] = ldsoff;                               // lds_addr
                d0[2] = (unsigned)ga;                         // global_addr[31:0]
                d0[3] = ((unsigned)(ga >> 32) & 0x01FFFFFFu)  // global_addr[56:32]
                        | 0x80000000u;                        // type=2 ("image")
                v8u d1;
                d1[0] = 0x00020000u;                          // data_size=2 (4B), no flags
                d1[1] = (td0 & 0xFFFFu) << 16;                // tensor_dim0[15:0]
                d1[2] = (td0 >> 16) | ((td1 & 0xFFFFu) << 16);// td0[31:16] | td1[15:0]
                d1[3] = (td1 >> 16) | (32u << 16);            // td1[31:16] | tile_dim0=32
                d1[4] = 64u;                                  // tile_dim1=64, tile_dim2=0
                d1[5] = (unsigned)K;                          // tensor_dim0_stride[31:0]
                d1[6] = 0u;                                   // stride hi / dim1_stride lo
                d1[7] = 0u;
                asm volatile("tensor_load_to_lds %0, %1" :: "s"(d0), "s"(d1) : "memory");
                __builtin_amdgcn_s_wait_tensorcnt(0);
            }
            if (k0 + 32 < K) __builtin_prefetch(w + (size_t)co0 * K + k0 + 32, 0, 1);
        } else {
            for (int e = tid; e < 64 * 32; e += 128) {
                const int n  = e >> 5;
                const int kk = e & 31;
                const int k  = k0 + kk;
                const int co = co0 + n;
                float v = 0.0f;
                if (k < K && co < Cout) {
                    const int ci  = k / KK;
                    const int rem = k - ci * KK;
                    v = w[(ci * Cout + co) * KK + rem];       // (Cin,Cout,kh,kw)
                }
                Wst[n][kk] = v;
            }
        }
        __syncthreads();

        // ---- convert weight chunk to f16 ----
        for (int e = tid; e < 64 * 32; e += 128)
            Bsh[e >> 5][e & 31] = (_Float16)Wst[e >> 5][e & 31];
        __syncthreads();

        // ---- fragments (ISA 16-bit A/B layout: two 8-half runs per lane) ----
        FragH fa;
        const int arow = wv * 16 + lm;
        fa.q[0] = *(const U128*)&Ash[arow][lh * 8];
        fa.q[1] = *(const U128*)&Ash[arow][16 + lh * 8];

        #pragma unroll
        for (int ct = 0; ct < 4; ++ct) {
            FragH fb;
            const int brow = ct * 16 + lm;
            fb.q[0] = *(const U128*)&Bsh[brow][lh * 8];
            fb.q[1] = *(const U128*)&Bsh[brow][16 + lh * 8];
            c[ct] = __builtin_amdgcn_wmma_f32_16x16x32_f16(false, fa.v, false, fb.v,
                                                           (short)0, c[ct], false, false);
        }
        __syncthreads();
    }

    // ---- store: lane holds column n, rows m = r + 8*lh ----
    #pragma unroll
    for (int ct = 0; ct < 4; ++ct) {
        const int co = co0 + ct * 16 + lm;
        if (co < Cout) {
            #pragma unroll
            for (int r = 0; r < 8; ++r) {
                const int m = r + 8 * lh;
                const int p = p0 + wv * 16 + m;
                if (p < HWo) y[(b * Cout + co) * HWo + p] = c[ct][r];
            }
        }
    }
}

// ---------------------------------------------------------------------------
// BatchNorm batch-stats: partial sums via LDS reduce + f32 global atomics.
// Grid (C, 16) so a 128 MB tensor is reduced by 1024 blocks, not 64.
// ---------------------------------------------------------------------------
__global__ void k_zero(float* p, int n) {
    const int i = blockIdx.x * blockDim.x + threadIdx.x;
    if (i < n) p[i] = 0.0f;
}

__global__ void k_bn_stats(const float* __restrict__ x, float* __restrict__ sum,
                           float* __restrict__ sq, int C, int HW) {
    __shared__ float s1[256], s2[256];
    const int cch = blockIdx.x;
    const int tid = threadIdx.x;
    const int cnt = BATCH * HW;
    const int step = 256 * gridDim.y;
    float a = 0.0f, b2 = 0.0f;
    for (int j = blockIdx.y * 256 + tid; j < cnt; j += step) {
        const int bb = j / HW;
        const int p  = j - bb * HW;
        const float v = x[(bb * C + cch) * HW + p];
        a += v; b2 += v * v;
    }
    s1[tid] = a; s2[tid] = b2;
    __syncthreads();
    for (int s = 128; s > 0; s >>= 1) {
        if (tid < s) { s1[tid] += s1[tid + s]; s2[tid] += s2[tid + s]; }
        __syncthreads();
    }
    if (tid == 0) {
        atomicAdd(&sum[cch], s1[0]);
        atomicAdd(&sq[cch],  s2[0]);
    }
}

// normalize + relu (+ optional residual add); res may alias out (same index).
__global__ void k_bn_apply(const float* __restrict__ x, const float* __restrict__ sum,
                           const float* __restrict__ sq, const float* res,
                           float* out, int C, int HW, int N) {
    const int i = blockIdx.x * blockDim.x + threadIdx.x;
    if (i >= N) return;
    const int cch = (i / HW) % C;
    const float inv = 1.0f / (float)(BATCH * HW);
    const float m  = sum[cch] * inv;
    const float vv = sq[cch] * inv - m * m;
    float v = (x[i] - m) * rsqrtf(vv + EPSBN);
    v = v > 0.0f ? v : 0.0f;
    if (res) v += res[i];
    out[i] = v;
}

__global__ void k_maxpool2(const float* __restrict__ x, float* __restrict__ y,
                           int C, int Ho, int Wo) {
    const int i = blockIdx.x * blockDim.x + threadIdx.x;
    const int N = BATCH * C * Ho * Wo;
    if (i >= N) return;
    const int ox = i % Wo;
    const int oy = (i / Wo) % Ho;
    const int bc = i / (Wo * Ho);
    const int Wi = 2 * Wo;
    const float* p = x + (bc * (2 * Ho) + 2 * oy) * Wi + 2 * ox;
    float m = p[0];
    m = fmaxf(m, p[1]);
    m = fmaxf(m, p[Wi]);
    m = fmaxf(m, p[Wi + 1]);
    y[i] = m;
}

__global__ void k_enorm(const float* __restrict__ E, float* __restrict__ en) {
    const int i = blockIdx.x * blockDim.x + threadIdx.x;
    if (i >= 512) return;
    float s = 0.0f;
    for (int k = 0; k < 64; ++k) { const float v = E[i * 64 + k]; s += v * v; }
    en[i] = s;
}

// ---------------------------------------------------------------------------
// VQ: per block (1 wave) 16 (b,h,w) rows; dot(z,E) via WMMA f16, K=64 (2 chunks),
// 32 code-tiles of 16; argmin via LDS; gather winning code rows into dec_in.
// ‖z‖² omitted (invariant under argmin over codes).
// ---------------------------------------------------------------------------
__global__ void k_vq(const float* __restrict__ enc, const float* __restrict__ E,
                     const float* __restrict__ enorm, float* __restrict__ dec_in) {
    __shared__ float distS[16][16];
    const int lane = threadIdx.x;
    const int lm   = lane & 15;
    const int lh   = lane >> 4;
    const int g0   = blockIdx.x * 16;

    FragH a0, a1;
    {
        const int g  = g0 + lm;
        const int bb = g >> 8;
        const int hw = g & 255;
        #pragma unroll
        for (int j = 0; j < 8; ++j) {
            const int ka = lh * 8 + j;
            const int kb = 16 + lh * 8 + j;
            a0.h[j]     = (_Float16)enc[(bb * 64 + ka) * 256 + hw];
            a0.h[8 + j] = (_Float16)enc[(bb * 64 + kb) * 256 + hw];
            a1.h[j]     = (_Float16)enc[(bb * 64 + 32 + ka) * 256 + hw];
            a1.h[8 + j] = (_Float16)enc[(bb * 64 + 32 + kb) * 256 + hw];
        }
    }

    float bestv = 3.0e38f;
    int   besti = 0;

    for (int t = 0; t < 32; ++t) {
        const int n0 = t * 16;
        FragH b0, b1;
        const int col = n0 + lm;
        #pragma unroll
        for (int j = 0; j < 8; ++j) {
            const int ka = lh * 8 + j;
            const int kb = 16 + lh * 8 + j;
            b0.h[j]     = (_Float16)E[col * 64 + ka];
            b0.h[8 + j] = (_Float16)E[col * 64 + kb];
            b1.h[j]     = (_Float16)E[col * 64 + 32 + ka];
            b1.h[8 + j] = (_Float16)E[col * 64 + 32 + kb];
        }
        v8f c;
        #pragma unroll
        for (int r = 0; r < 8; ++r) c[r] = 0.0f;
        c = __builtin_amdgcn_wmma_f32_16x16x32_f16(false, a0.v, false, b0.v,
                                                   (short)0, c, false, false);
        c = __builtin_amdgcn_wmma_f32_16x16x32_f16(false, a1.v, false, b1.v,
                                                   (short)0, c, false, false);
        #pragma unroll
        for (int r = 0; r < 8; ++r) distS[r + 8 * lh][lm] = c[r];
        __syncthreads();
        if (lane < 16) {
            for (int n = 0; n < 16; ++n) {
                const float d = enorm[n0 + n] - 2.0f * distS[lane][n];
                if (d < bestv) { bestv = d; besti = n0 + n; }
            }
        }
        __syncthreads();
    }

    if (lane < 16) {
        const int g  = g0 + lane;
        const int bb = g >> 8;
        const int hw = g & 255;
        for (int cch = 0; cch < 64; ++cch)
            dec_in[(bb * 64 + cch) * 256 + hw] = E[besti * 64 + cch];
    }
}

// ---------------------------------------------------------------------------
// Host-side helpers / launch sequence
// ---------------------------------------------------------------------------
static void run_cblock(hipStream_t s, const float* x, const float* w, const float* bias,
                       float* tmp, float* stat, float* out, const float* res,
                       int Cin, int Cout, int KH, int KW, int pad,
                       int Hin, int Win, int Hout, int Wout, int tmode) {
    const int HW = Hout * Wout;
    dim3 g((HW + 63) / 64, (Cout + 63) / 64, BATCH);
    k_conv_wmma<<<g, 128, 0, s>>>(x, w, bias, tmp, Cin, Cout, KH, KW, pad,
                                  Hin, Win, Hout, Wout, tmode);
    k_zero<<<1, 128, 0, s>>>(stat, 128);
    k_bn_stats<<<dim3(Cout, 16), 256, 0, s>>>(tmp, stat, stat + 64, Cout, HW);
    const int N = BATCH * Cout * HW;
    k_bn_apply<<<(N + 255) / 256, 256, 0, s>>>(tmp, stat, stat + 64, res, out, Cout, HW, N);
}

static void run_maxpool(hipStream_t s, const float* x, float* y, int C, int Ho, int Wo) {
    const int N = BATCH * C * Ho * Wo;
    k_maxpool2<<<(N + 255) / 256, 256, 0, s>>>(x, y, C, Ho, Wo);
}

extern "C" void kernel_launch(void* const* d_in, const int* in_sizes, int n_in,
                              void* d_out, int out_size, void* d_ws, size_t ws_size,
                              hipStream_t stream) {
    (void)in_sizes; (void)n_in; (void)out_size; (void)ws_size;
    const float* X = (const float*)d_in[0];
#define P(i) ((const float*)d_in[i])

    float* ws    = (float*)d_ws;
    float* buf0  = ws;                       // 33,554,432 f (128 MB): widest conv outs
    float* buf2  = ws + 33554432;            // 33,554,432 f (128 MB)
    float* buf1  = ws + 67108864;            // 16,777,216 f (64 MB)
    float* stat  = ws + 83886080;            // sum[64] | sumsq[64]
    float* enorm = stat + 128;               // 512 f

    float* out    = (float*)d_out;
    float* enc    = out;                     // [32,64,16,16]
    float* dec_in = out + 524288;            // [32,64,16,16]
    float* dec    = out + 1048576;           // [32,1,128,128]

    // ---- Encoder ----
    run_cblock(stream, X,    P(1),  P(2),  buf0, stat, buf1, nullptr, 1, 32, 5, 5, 2, 128,128,128,128, 0);
    run_cblock(stream, buf1, P(3),  P(4),  buf0, stat, buf2, nullptr, 32,64, 5, 5, 2, 128,128,128,128, 0);
    run_cblock(stream, buf2, P(5),  P(6),  buf0, stat, buf2, buf1,    64,32, 5, 5, 2, 128,128,128,128, 0);
    run_maxpool(stream, buf2, buf1, 32, 64, 64);
    run_cblock(stream, buf1, P(7),  P(8),  buf0, stat, buf2, nullptr, 32,64, 3, 3, 1, 64,64,64,64, 0);
    run_cblock(stream, buf2, P(9),  P(10), buf0, stat, buf1, nullptr, 64,64, 3, 3, 1, 64,64,64,64, 0);
    run_cblock(stream, buf1, P(11), P(12), buf0, stat, buf2, buf2,    64,64, 3, 3, 1, 64,64,64,64, 0);
    run_maxpool(stream, buf2, buf1, 64, 32, 32);
    run_cblock(stream, buf1, P(13), P(14), buf0, stat, buf2, nullptr, 64,64, 1, 1, 0, 32,32,32,32, 0);
    run_cblock(stream, buf2, P(15), P(16), buf0, stat, buf2, buf1,    64,64, 1, 1, 0, 32,32,32,32, 0);
    run_maxpool(stream, buf2, enc, 64, 16, 16);   // enc -> d_out region 0

    // ---- Vector quantization ----
    k_enorm<<<2, 256, 0, stream>>>(P(17), enorm);
    k_vq<<<512, 32, 0, stream>>>(enc, P(17), enorm, dec_in);   // dec_in -> d_out region 1

    // ---- Decoder ----
    run_cblock(stream, dec_in, P(18), P(19), buf0, stat, buf1, nullptr, 64,32, 4, 4, 1, 16,16,32,32, 1);
    run_cblock(stream, buf1,   P(20), P(21), buf0, stat, buf2, nullptr, 32,64, 3, 3, 1, 32,32,32,32, 0);
    run_cblock(stream, buf2,   P(22), P(23), buf0, stat, buf1, buf1,    64,32, 3, 3, 1, 32,32,32,32, 0);
    run_cblock(stream, buf1,   P(24), P(25), buf0, stat, buf2, nullptr, 32,16, 4, 4, 1, 32,32,64,64, 1);
    run_cblock(stream, buf2,   P(26), P(27), buf0, stat, buf1, nullptr, 16,64, 3, 3, 1, 64,64,64,64, 0);
    run_cblock(stream, buf1,   P(28), P(29), buf0, stat, buf2, buf2,    64,16, 3, 3, 1, 64,64,64,64, 0);
    run_cblock(stream, buf2,   P(30), P(31), buf0, stat, buf1, nullptr, 16, 1, 4, 4, 1, 64,64,128,128, 1);
    run_cblock(stream, buf1,   P(32), P(33), buf0, stat, buf2, nullptr, 1, 64, 1, 1, 0, 128,128,128,128, 0);
    run_cblock(stream, buf2,   P(34), P(35), buf0, stat, dec,  buf1,    64, 1, 1, 1, 0, 128,128,128,128, 0);
#undef P
}
